// GCN_edge_conditional_54039278519098
// MI455X (gfx1250) — matrix operand order repaired and checked by reading, not compiled, not run on hardware
//
#include <hip/hip_runtime.h>

typedef __attribute__((ext_vector_type(16))) _Float16 v16h;
typedef __attribute__((ext_vector_type(8)))  _Float16 v8h;
typedef __attribute__((ext_vector_type(8)))  float    v8f;

#define NNODES 8192
#define NGRAPH 64
#define NPERG  128
#define NEDGE0 32768
#define NETOT  (NEDGE0 + NNODES)   // 40960 (self-loops appended)
#define IN_F   32
#define HOUT   64
#define BOND   5
#define K1     (6 * IN_F)          // 192  (augmented edge dim 6 x din)
#define K2     (6 * HOUT)          // 384

// ---------------------------------------------------------------------------
// prep: build transposed f16 fused-weight matrices Tt[o][k] (k = b*din + i),
// where b<5 -> edge_W[b, i*64+o], b==5 -> edge_b[i*64+o]; and bias-init aggs.
// ---------------------------------------------------------------------------
__global__ void prep_kernel(const float* __restrict__ eW1, const float* __restrict__ eb1,
                            const float* __restrict__ b1,
                            const float* __restrict__ eW2, const float* __restrict__ eb2,
                            const float* __restrict__ b2,
                            _Float16* __restrict__ T1t, _Float16* __restrict__ T2t,
                            float* __restrict__ agg1, float* __restrict__ agg2) {
  const int nT1 = K1 * HOUT;
  const int nT2 = K2 * HOUT;
  const int nAg = NNODES * HOUT;
  const int total = nT1 + nT2 + 2 * nAg;
  for (int idx = blockIdx.x * blockDim.x + threadIdx.x; idx < total;
       idx += gridDim.x * blockDim.x) {
    if (idx < nT1) {
      int o = idx / K1, k = idx % K1;
      int b = k / IN_F, i = k % IN_F;
      float v = (b < BOND) ? eW1[b * (IN_F * HOUT) + i * HOUT + o] : eb1[i * HOUT + o];
      T1t[idx] = (_Float16)v;
    } else if (idx < nT1 + nT2) {
      int t = idx - nT1;
      int o = t / K2, k = t % K2;
      int b = k / HOUT, i = k % HOUT;
      float v = (b < BOND) ? eW2[b * (HOUT * HOUT) + i * HOUT + o] : eb2[i * HOUT + o];
      T2t[t] = (_Float16)v;
    } else if (idx < nT1 + nT2 + nAg) {
      int t = idx - nT1 - nT2;
      agg1[t] = b1[t % HOUT];
    } else {
      int t = idx - nT1 - nT2 - nAg;
      agg2[t] = b2[t % HOUT];
    }
  }
}

// ---------------------------------------------------------------------------
// Fused NNConv layer as dense GEMM: A[16 edges x K] (K = 6*DIN, built as
// p(edge) (x) h[src]) times Tt, D scattered to agg[dst] via f32 atomics.
// One wave per 16-edge tile, 4 waves per block. 4 WMMA accumulators cover 64
// output channels.
// ---------------------------------------------------------------------------
template <int DIN, bool RELU_IN>
__global__ __launch_bounds__(128) void nnconv_kernel(
    const float* __restrict__ hsrc,          // [N, DIN] (pre-ReLU if RELU_IN)
    const float* __restrict__ ef,            // [E0, BOND]
    const int* __restrict__ src, const int* __restrict__ dst,
    const _Float16* __restrict__ Tt,         // [64, K] transposed fused weights
    float* __restrict__ agg)                 // [N, 64], bias pre-loaded
{
  constexpr int K = 6 * DIN;
  __shared__ _Float16 Arows[4][16][K];       // per-wave A staging (<= 48 KB)

  const int wave = threadIdx.x >> 5;
  const int lane = threadIdx.x & 31;
  const int tile = blockIdx.x * 4 + wave;    // 2560 tiles exactly
  const int Hs   = lane >> 4;                // lane half-select
  const int nb   = lane & 15;

  // --- stage A rows: row m = outer(p[e], h[src[e]]) in f16 ---
  for (int m = 0; m < 16; ++m) {
    int e = tile * 16 + m;
    bool real = (e < NEDGE0);
    int s = real ? src[e] : (e - NEDGE0);
    const float* hrow = hsrc + (size_t)s * DIN;
    for (int k = lane; k < K; k += 32) {
      int b = k / DIN;
      int i = k - b * DIN;
      float hv = hrow[i];
      if (RELU_IN) hv = fmaxf(hv, 0.f);
      float pv = (b == BOND) ? 1.f : (real ? ef[(size_t)e * BOND + b] : 0.f);
      Arows[wave][m][k] = (_Float16)(pv * hv);
    }
  }
  __syncthreads();

  v8f c0 = {}, c1 = {}, c2 = {}, c3 = {};
  const _Float16* arow = &Arows[wave][nb][0];

  union AU { v16h v; v8h h[2]; };

  for (int ks = 0; ks < K; ks += 32) {
    // A fragment: lanes 0-15 hold K {0..15}, lanes 16-31 {16..31}
    AU au;
    au.h[0] = *(const v8h*)(arow + ks + 16 * Hs);
    au.h[1] = *(const v8h*)(arow + ks + 16 * Hs + 8);
    v16h a = au.v;
    // B fragments: column N = nb + 16*t, K contiguous (transposed T) ->
    // one 32-byte load each.
    const _Float16* tb = Tt + ks + 16 * Hs;
    v16h b0 = *(const v16h*)(tb + (size_t)(nb +  0) * K);
    v16h b1 = *(const v16h*)(tb + (size_t)(nb + 16) * K);
    v16h b2 = *(const v16h*)(tb + (size_t)(nb + 32) * K);
    v16h b3 = *(const v16h*)(tb + (size_t)(nb + 48) * K);
    c0 = __builtin_amdgcn_wmma_f32_16x16x32_f16(false, a, false, b0, (short)0, c0, false, false);
    c1 = __builtin_amdgcn_wmma_f32_16x16x32_f16(false, a, false, b1, (short)0, c1, false, false);
    c2 = __builtin_amdgcn_wmma_f32_16x16x32_f16(false, a, false, b2, (short)0, c2, false, false);
    c3 = __builtin_amdgcn_wmma_f32_16x16x32_f16(false, a, false, b3, (short)0, c3, false, false);
  }

  // --- scatter-add: D element (VGPR r, lane) = edge M = r + 8*Hs, chan = t*16+nb
  int dIdx[8];
#pragma unroll
  for (int r = 0; r < 8; ++r) {
    int e = tile * 16 + (r + 8 * Hs);
    dIdx[r] = (e < NEDGE0) ? dst[e] : (e - NEDGE0);
  }
  v8f cc[4] = {c0, c1, c2, c3};
#pragma unroll
  for (int t = 0; t < 4; ++t) {
#pragma unroll
    for (int r = 0; r < 8; ++r) {
      float val = cc[t][r];
      atomicAdd(&agg[(size_t)dIdx[r] * HOUT + t * 16 + nb], val);
    }
  }
}

// ---------------------------------------------------------------------------
// Per-graph pooling + timestep conditioning. One 64-thread block per graph.
// h = relu(agg2) (bias already folded in). out[g] = tanh(relu([wsum|max]+pe)).
// ---------------------------------------------------------------------------
__global__ __launch_bounds__(64) void pool_kernel(
    const float* __restrict__ agg2, const float* __restrict__ wsW,
    const float* __restrict__ wsb, const float* __restrict__ ts,
    float* __restrict__ out) {
  int g = blockIdx.x;
  int o = threadIdx.x;  // 0..63
  __shared__ float wn[NPERG];
  __shared__ float wl[HOUT];
  wl[o] = wsW[o];
  __syncthreads();
  // node gating weights: sigmoid(h . wsW + b)
  for (int rep = 0; rep < 2; ++rep) {
    int n = o + rep * 64;
    const float* row = agg2 + ((size_t)g * NPERG + n) * HOUT;
    float acc = wsb[0];
    for (int i = 0; i < HOUT; ++i) acc += fmaxf(row[i], 0.f) * wl[i];
    wn[n] = 1.f / (1.f + __expf(-acc));
  }
  __syncthreads();
  float s = 0.f, mx = -INFINITY;
  for (int n = 0; n < NPERG; ++n) {
    float h = fmaxf(agg2[((size_t)g * NPERG + n) * HOUT + o], 0.f);
    s += h * wn[n];
    mx = fmaxf(mx, h);
  }
  float t = ts[g];
  // inv_freq[o] = 10000^(-(2o)/128)
  float inv = __expf(-((float)(2 * o) / 128.f) * 9.210340371976184f);
  float cs = s + __sinf(t * inv);   // channel o      (weighted-sum half)
  float cm = mx + __cosf(t * inv);  // channel 64+o   (max half)
  out[g * 128 + o]      = tanhf(fmaxf(cs, 0.f));
  out[g * 128 + 64 + o] = tanhf(fmaxf(cm, 0.f));
}

// ---------------------------------------------------------------------------
extern "C" void kernel_launch(void* const* d_in, const int* in_sizes, int n_in,
                              void* d_out, int out_size, void* d_ws, size_t ws_size,
                              hipStream_t stream) {
  const float* node_feats = (const float*)d_in[0];
  const float* edge_feats = (const float*)d_in[1];
  const int*   src        = (const int*)d_in[2];
  const int*   dst        = (const int*)d_in[3];
  /* d_in[4] graph_ids: contiguous blocks of 128, mapping is implicit */
  const float* timestep   = (const float*)d_in[5];
  const float* eW1 = (const float*)d_in[6];
  const float* eb1 = (const float*)d_in[7];
  const float* b1  = (const float*)d_in[8];
  const float* eW2 = (const float*)d_in[9];
  const float* eb2 = (const float*)d_in[10];
  const float* b2  = (const float*)d_in[11];
  const float* wsW = (const float*)d_in[12];
  const float* wsb = (const float*)d_in[13];
  float* out = (float*)d_out;

  char* ws = (char*)d_ws;
  const size_t aggBytes = (size_t)NNODES * HOUT * sizeof(float);  // 2 MB each
  float*    agg1 = (float*)(ws);
  float*    agg2 = (float*)(ws + aggBytes);
  _Float16* T1t  = (_Float16*)(ws + 2 * aggBytes);
  _Float16* T2t  = (_Float16*)(ws + 2 * aggBytes + (size_t)K1 * HOUT * sizeof(_Float16));

  prep_kernel<<<256, 256, 0, stream>>>(eW1, eb1, b1, eW2, eb2, b2, T1t, T2t, agg1, agg2);
  // 2560 edge tiles = 640 blocks x 4 waves (exact)
  nnconv_kernel<IN_F, false><<<640, 128, 0, stream>>>(node_feats, edge_feats, src, dst, T1t, agg1);
  nnconv_kernel<HOUT, true ><<<640, 128, 0, stream>>>(agg1,       edge_feats, src, dst, T2t, agg2);
  pool_kernel<<<NGRAPH, 64, 0, stream>>>(agg2, wsW, wsb, timestep, out);
}
